// Transformer_super_V3_89670327206367
// MI455X (gfx1250) — matrix operand order repaired and checked by reading, not compiled, not run on hardware
//
#include <hip/hip_runtime.h>

// ---------------------------------------------------------------------------
// CDNA5 (gfx1250) fused pairwise-bilinear + tanh + projection kernel.
//   h[p] = tanh(xa_p @ Wa_p^T + xb_p @ Wb_p^T + b1_p)      (15 pairs)
//   out  = concat_p(h[p]) @ Wf^T + bf
// fp32 -> bf16 pre-convert, then one fused kernel using
// v_wmma_f32_16x16x32_bf16 with the stage-1 result staged through LDS.
// Biases are folded into the WMMA accumulator init; tanh uses the
// hardware exp2/rcp transcendentals (co-executes with the matrix pipe).
// ---------------------------------------------------------------------------

typedef __attribute__((ext_vector_type(16))) __bf16 v16bf;
typedef __attribute__((ext_vector_type(8)))  float  v8f;

#define NCF      6
#define P_PAIRS  15
#define BS       16384          // B*S = 32*512
#define Ddim     256
#define PD       (P_PAIRS * Ddim)   // 3840
#define HPAD     264            // LDS row stride in bf16 elems (conflict-free)

__constant__ int c_ia[P_PAIRS] = {0,0,0,0,0,1,1,1,1,2,2,2,3,3,4};
__constant__ int c_ib[P_PAIRS] = {1,2,3,4,5,2,3,4,5,3,4,5,4,5,5};

__device__ __forceinline__ unsigned short f2bf(float f) {
    unsigned int u = __float_as_uint(f);
    u += 0x7FFFu + ((u >> 16) & 1u);          // round-to-nearest-even
    return (unsigned short)(u >> 16);
}

// tanh(x) = 1 - 2/(e^(2x)+1), via v_exp_f32 / v_rcp_f32 (~5 VALU/TRANS ops).
// exp2 saturates correctly for large |x| (-> +/-1), so no clamping needed.
__device__ __forceinline__ float fast_tanh(float x) {
    float e = __builtin_amdgcn_exp2f(x * 2.8853900817779268f);  // 2*log2(e)
    float r = __builtin_amdgcn_rcpf(e + 1.0f);
    return __builtin_fmaf(-2.0f, r, 1.0f);
}

__device__ __forceinline__ v8f splat8(float s) {
    v8f z = {s, s, s, s, s, s, s, s};
    return z;
}

union FragU { v16bf bf; uint4 q[2]; };

// A-fragment (16x32 bf16): lane L holds row M=L%16; half h=L/16 selects
// k-chunks [h*8, h*8+7] and [16+h*8, 16+h*8+7]. base points at (row0, k0).
__device__ __forceinline__ v16bf loadA(const unsigned short* base, int stride, int lane) {
    const int m = lane & 15, h = lane >> 4;
    const unsigned short* r = base + (size_t)m * stride + h * 8;
    FragU f;
    f.q[0] = *(const uint4*)(r);
    f.q[1] = *(const uint4*)(r + 16);
    return f.bf;
}

// B-fragment (32x16 bf16) built from row-major W so that B[k][n] = W[n][k]:
// lane L holds col N=L%16; half h=L/16 selects k-chunks [h*8, h*8+7] and [16+h*8, 16+h*8+7].
// base points at W element (n0, k0); stride = W row length in elems.
__device__ __forceinline__ v16bf loadB(const unsigned short* base, int stride, int lane) {
    const int n = lane & 15, h = lane >> 4;
    const unsigned short* r = base + (size_t)n * stride + h * 8;
    FragU f;
    f.q[0] = *(const uint4*)(r);
    f.q[1] = *(const uint4*)(r + 16);
    return f.bf;
}

// ---------------------------------------------------------------------------
__global__ void convert_f32_to_bf16(const float* __restrict__ src,
                                    unsigned short* __restrict__ dst, int n) {
    int i = (blockIdx.x * blockDim.x + threadIdx.x) * 4;
    if (i + 3 < n) {
        float4 v = *(const float4*)(src + i);
        uint2 o;
        o.x = (unsigned int)f2bf(v.x) | ((unsigned int)f2bf(v.y) << 16);
        o.y = (unsigned int)f2bf(v.z) | ((unsigned int)f2bf(v.w) << 16);
        *(uint2*)(dst + i) = o;
    }
}

// ---------------------------------------------------------------------------
// Fused kernel: 256 threads (8 wave32s). WG tile = 64 rows x 256 cols of out.
// Wave (r,c): r = wave&3 row group (16 rows), c = wave>>2 col group (128 cols).
__global__ __launch_bounds__(256)
void fused_pair_mlp(const unsigned short* __restrict__ fb,   // bf16 features [6][16384][256]
                    const unsigned short* __restrict__ w1b,  // bf16 W1 [15][256][512]
                    const unsigned short* __restrict__ wfb,  // bf16 Wf [256][3840]
                    const float* __restrict__ b1,            // f32 [15][256]
                    const float* __restrict__ bfv,           // f32 [256]
                    float* __restrict__ out)                 // f32 [16384][256]
{
    __shared__ unsigned short hlds[64 * HPAD];   // 33,792 B: bf16 h tile (64x256, padded)

    const int lane    = threadIdx.x & 31;
    const int wave    = threadIdx.x >> 5;
    const int rLocal  = (wave & 3) * 16;     // wave's row offset in WG tile
    const int cBase   = (wave >> 2) * 128;   // wave's col offset (0 or 128)
    const int rowBase = blockIdx.x * 64;     // WG's row offset in [0,16384)

    const int halfsel = lane >> 4;           // C/D layout: upper half-wave = rows +8
    const int nLane   = lane & 15;           // C/D layout: lane selects column

    // out accumulators, pre-seeded with the final bias bf (folded epilogue add)
    v8f oacc[8];
    #pragma unroll
    for (int t = 0; t < 8; ++t) oacc[t] = splat8(bfv[cBase + t * 16 + nLane]);

    for (int p = 0; p < P_PAIRS; ++p) {
        const int fa  = c_ia[p];
        const int fbx = c_ib[p];
        const unsigned short* xa = fb + ((size_t)fa  * BS + rowBase + rLocal) * Ddim;
        const unsigned short* xb = fb + ((size_t)fbx * BS + rowBase + rLocal) * Ddim;
        const unsigned short* wp = w1b + (size_t)p * Ddim * 512;

        if (p + 1 < P_PAIRS) {   // global_prefetch_b8 for next pair's weights
            __builtin_prefetch(w1b + (size_t)(p + 1) * Ddim * 512, 0, 1);
            __builtin_prefetch(wfb + (size_t)(p + 1) * Ddim, 0, 1);
        }

        // ---- Stage 1: h(16x128) = b1 + xa@Wa^T + xb@Wb^T (K=256, bf16 WMMA) ----
        // b1 folded into accumulator init: C layout holds one column per lane.
        v8f hacc[8];
        #pragma unroll
        for (int t = 0; t < 8; ++t) hacc[t] = splat8(b1[p * Ddim + cBase + t * 16 + nLane]);

        for (int k0 = 0; k0 < 256; k0 += 32) {
            v16bf Aa = loadA(xa + k0, Ddim, lane);
            v16bf Ab = loadA(xb + k0, Ddim, lane);
            #pragma unroll
            for (int t = 0; t < 8; ++t) {
                const unsigned short* wrow = wp + (size_t)(cBase + t * 16) * 512 + k0;
                v16bf Ba = loadB(wrow,        512, lane);   // Wa: cols 0..255
                v16bf Bb = loadB(wrow + 256,  512, lane);   // Wb: cols 256..511
                hacc[t] = __builtin_amdgcn_wmma_f32_16x16x32_bf16(
                              false, Aa, false, Ba, (short)0, hacc[t], false, false);
                hacc[t] = __builtin_amdgcn_wmma_f32_16x16x32_bf16(
                              false, Ab, false, Bb, (short)0, hacc[t], false, false);
            }
        }

        // fast tanh -> bf16 -> padded LDS tile
        #pragma unroll
        for (int t = 0; t < 8; ++t) {
            const int col = cBase + t * 16 + nLane;
            const int rb  = rLocal + 8 * halfsel;
            #pragma unroll
            for (int v = 0; v < 8; ++v) {
                hlds[(rb + v) * HPAD + col] = f2bf(fast_tanh(hacc[t][v]));
            }
        }
        __syncthreads();

        // ---- Stage 2: out(16x128) += h(16x256) @ Wf_p^T (K = 256) ----
        for (int k0 = 0; k0 < 256; k0 += 32) {
            v16bf Ah = loadA(&hlds[rLocal * HPAD + k0], HPAD, lane);  // ds_load_b128
            #pragma unroll
            for (int t = 0; t < 8; ++t) {
                const unsigned short* wf =
                    wfb + (size_t)(cBase + t * 16) * PD + p * Ddim + k0;
                v16bf Bf = loadB(wf, PD, lane);
                oacc[t] = __builtin_amdgcn_wmma_f32_16x16x32_bf16(
                              false, Ah, false, Bf, (short)0, oacc[t], false, false);
            }
        }
        __syncthreads();   // hlds reused next pair
    }

    // ---- Epilogue: store fp32 (bf already folded into oacc init) ----
    #pragma unroll
    for (int t = 0; t < 8; ++t) {
        const int col = cBase + t * 16 + nLane;
        const int rb  = rowBase + rLocal + 8 * halfsel;
        #pragma unroll
        for (int v = 0; v < 8; ++v) {
            out[(size_t)(rb + v) * Ddim + col] = oacc[t][v];
        }
    }
}

// ---------------------------------------------------------------------------
extern "C" void kernel_launch(void* const* d_in, const int* in_sizes, int n_in,
                              void* d_out, int out_size, void* d_ws, size_t ws_size,
                              hipStream_t stream)
{
    const float* features = (const float*)d_in[0];   // [6][32][512][256]
    const float* W1       = (const float*)d_in[1];   // [15][256][512]
    const float* b1       = (const float*)d_in[2];   // [15][256]
    const float* Wf       = (const float*)d_in[3];   // [256][3840]
    const float* bf       = (const float*)d_in[4];   // [256]
    float* out            = (float*)d_out;           // [16384][256]

    const int nF  = NCF * BS * Ddim;        // 25,165,824
    const int nW1 = P_PAIRS * Ddim * 512;   //  1,966,080
    const int nWf = Ddim * PD;              //    983,040

    unsigned short* fb  = (unsigned short*)d_ws;
    unsigned short* w1b = fb  + (size_t)nF;
    unsigned short* wfb = w1b + (size_t)nW1;

    // bf16 pre-conversion (coalesced, 4 elems/thread; all sizes divisible by 1024)
    convert_f32_to_bf16<<<nF  / 1024, 256, 0, stream>>>(features, fb,  nF);
    convert_f32_to_bf16<<<nW1 / 1024, 256, 0, stream>>>(W1,       w1b, nW1);
    convert_f32_to_bf16<<<nWf / 1024, 256, 0, stream>>>(Wf,       wfb, nWf);

    // fused WMMA kernel: 16384 rows / 64 rows per WG = 256 workgroups
    fused_pair_mlp<<<BS / 64, 256, 0, stream>>>(fb, w1b, wfb, b1, bf, out);
}
